// MoE_77644418777543
// MI455X (gfx1250) — compile-verified, matrix operand order checked by reading, not run on hardware
//
#include <hip/hip_runtime.h>
#include <hip/hip_bf16.h>
#include <stdint.h>

typedef __attribute__((ext_vector_type(16))) _Float16 v16h;
typedef __attribute__((ext_vector_type(8)))  _Float16 v8h;
typedef __attribute__((ext_vector_type(8)))  float    v8f;

#define BM 128
#define BN 128
#define BK 32
#define LDS_STRIDE 40   // halfs; 80B row stride, keeps 16B alignment, spreads banks

#define EPI_BIAS_F32      0
#define EPI_BN_GELU_F16   1
#define EPI_BN_GELU_WACC  2

__device__ __forceinline__ float gelu_exact(float x) {
    return 0.5f * x * (1.0f + erff(x * 0.70710678118654752440f));
}

// Load a 16x32 (rows x K) f16 fragment from LDS into the WMMA A/B register layout.
// ISA 16-bit A/B layout: lanes 0-15 -> row = base+lane, halves 0..7 = K0..7, 8..15 = K16..23
//                        lanes 16-31 -> row = base+lane-16, halves 0..7 = K8..15, 8..15 = K24..31
__device__ __forceinline__ v16h load_frag(const _Float16* smem, int rowBase, int lane) {
    int l15 = lane & 15;
    int lh  = (lane >> 4) & 1;
    const _Float16* p = smem + (size_t)(rowBase + l15) * LDS_STRIDE + lh * 8;
    v8h lo = *(const v8h*)(p);
    v8h hi = *(const v8h*)(p + 16);
    v16h f;
#pragma unroll
    for (int i = 0; i < 8; ++i) { f[i] = lo[i]; f[i + 8] = hi[i]; }
    return f;
}

// ---------------------------------------------------------------------------
// Tiled WMMA GEMM:  C[M,N] = A[M,K] (f16, row-major) * Bt[N,K]^T (f16, W^T layout)
// Block 128x128, 8 waves, wave tile 32x64 (2x4 WMMA accumulators).
// Double-buffered LDS; global->VGPR loads for tile s+1 overlap WMMAs on tile s;
// global_prefetch for tile s+2. M%128==0, N%128==0, K%32==0 (guaranteed here).
// ---------------------------------------------------------------------------
__global__ __launch_bounds__(256)
void gemm_wmma_kernel(const _Float16* __restrict__ A, const _Float16* __restrict__ Bt,
                      int M, int N, int K, int epi,
                      const float* __restrict__ bias,
                      const float* __restrict__ bn_g, const float* __restrict__ bn_b,
                      const float* __restrict__ bn_m, const float* __restrict__ bn_v,
                      float* __restrict__ outF32, _Float16* __restrict__ outF16,
                      const float* __restrict__ wsel, const int* __restrict__ isel,
                      int expert, int accum_first)
{
    __shared__ _Float16 As[2][BM * LDS_STRIDE];
    __shared__ _Float16 Bs[2][BN * LDS_STRIDE];

    const int tid  = threadIdx.x;
    const int lane = tid & 31;
    const int wid  = tid >> 5;     // 0..7
    const int wm   = wid >> 1;     // 0..3  (row group of 32)
    const int wn   = wid & 1;      // 0..1  (col group of 64)
    const int row0 = blockIdx.y * BM;
    const int col0 = blockIdx.x * BN;

    // staging chunk assignment: 128 rows x 32 halfs = 512 chunks of 8 halfs,
    // 2 chunks per thread for each of A and B.
    int srow[2], soff[2];
#pragma unroll
    for (int it = 0; it < 2; ++it) {
        int c = tid + it * 256;
        srow[it] = c >> 2;
        soff[it] = (c & 3) * 8;
    }

    v8h ra[2], rb[2];
    auto gload = [&](int k0) {
#pragma unroll
        for (int it = 0; it < 2; ++it) {
            const _Float16* pa = A  + (size_t)(row0 + srow[it]) * K + k0 + soff[it];
            const _Float16* pb = Bt + (size_t)(col0 + srow[it]) * K + k0 + soff[it];
            ra[it] = *(const v8h*)pa;
            rb[it] = *(const v8h*)pb;
            __builtin_prefetch(pa + BK, 0, 1);   // -> global_prefetch_b8 (tile s+2)
            __builtin_prefetch(pb + BK, 0, 1);
        }
    };
    auto sstore = [&](int buf) {
#pragma unroll
        for (int it = 0; it < 2; ++it) {
            *(v8h*)&As[buf][(size_t)srow[it] * LDS_STRIDE + soff[it]] = ra[it];
            *(v8h*)&Bs[buf][(size_t)srow[it] * LDS_STRIDE + soff[it]] = rb[it];
        }
    };

    v8f acc[2][4] = {};

    const int nsteps = K / BK;
    gload(0);
    sstore(0);
    __syncthreads();

    for (int s = 0; s < nsteps; ++s) {
        const int cur = s & 1;
        if (s + 1 < nsteps) gload((s + 1) * BK);

        v16h af[2], bf[4];
#pragma unroll
        for (int mt = 0; mt < 2; ++mt) af[mt] = load_frag(As[cur], wm * 32 + mt * 16, lane);
#pragma unroll
        for (int nt = 0; nt < 4; ++nt) bf[nt] = load_frag(Bs[cur], wn * 64 + nt * 16, lane);

#pragma unroll
        for (int mt = 0; mt < 2; ++mt)
#pragma unroll
            for (int nt = 0; nt < 4; ++nt)
                acc[mt][nt] = __builtin_amdgcn_wmma_f32_16x16x32_f16(
                    false, af[mt], false, bf[nt], (short)0, acc[mt][nt], false, false);

        if (s + 1 < nsteps) sstore(cur ^ 1);
        __syncthreads();
    }

    // Epilogue. C/D layout: VGPR r: lanes 0-15 -> M=r, lanes 16-31 -> M=r+8; N = lane&15.
    const int l15 = lane & 15;
    const int lh  = (lane >> 4) & 1;
#pragma unroll
    for (int mt = 0; mt < 2; ++mt) {
#pragma unroll
        for (int nt = 0; nt < 4; ++nt) {
#pragma unroll
            for (int r = 0; r < 8; ++r) {
                int m = row0 + wm * 32 + mt * 16 + r + lh * 8;
                int n = col0 + wn * 64 + nt * 16 + l15;
                float x = acc[mt][nt][r] + bias[n];
                if (epi == EPI_BIAS_F32) {
                    outF32[(size_t)m * N + n] = x;
                    continue;
                }
                x = (x - bn_m[n]) * rsqrtf(bn_v[n] + 1e-5f) * bn_g[n] + bn_b[n];
                x = gelu_exact(x);
                if (epi == EPI_BN_GELU_F16) {
                    outF16[(size_t)m * N + n] = (_Float16)x;
                } else {
                    float w = 0.f;
#pragma unroll
                    for (int kk = 0; kk < 4; ++kk)
                        if (isel[m * 4 + kk] == expert) w += wsel[m * 4 + kk];
                    float val = w * x;
                    size_t o = (size_t)m * N + n;
                    if (accum_first) outF32[o] = val;
                    else             outF32[o] += val;
                }
            }
        }
    }
}

// ---------------------------------------------------------------------------
// Small helper kernels
// ---------------------------------------------------------------------------
__global__ void concat_f16_kernel(const float* __restrict__ wifi, const float* __restrict__ rfid,
                                  _Float16* __restrict__ out, int B) {
    int i = blockIdx.x * 256 + threadIdx.x;
    if (i >= B * 512) return;
    int b = i >> 9, c = i & 511;
    float v = (c < 256) ? wifi[b * 256 + c] : rfid[b * 256 + (c - 256)];
    out[i] = (_Float16)v;
}

// W[e][K][N] fp32 -> Wt[e][N][K] f16  (K contiguous so GEMM B-tiles load as b128)
__global__ void transpose_f16_kernel(const float* __restrict__ W, _Float16* __restrict__ Wt,
                                     int K, int N, int nmat) {
    size_t total = (size_t)nmat * K * N;
    size_t i = (size_t)blockIdx.x * 256 + threadIdx.x;
    if (i >= total) return;
    size_t per = (size_t)K * N;
    size_t e = i / per;
    size_t rem = i - e * per;
    int n = (int)(rem / K);
    int k = (int)(rem - (size_t)n * K);
    Wt[i] = (_Float16)W[e * per + (size_t)k * N + n];
}

__global__ void f32_to_f16_kernel(const float* __restrict__ in, _Float16* __restrict__ out, size_t n) {
    size_t i = (size_t)blockIdx.x * 256 + threadIdx.x;
    if (i < n) out[i] = (_Float16)in[i];
}

// Row-wise LayerNorm + exact GELU, fp32 in -> f16 out. One block per row.
__global__ __launch_bounds__(256)
void ln_gelu_kernel(const float* __restrict__ X, const float* __restrict__ g,
                    const float* __restrict__ bta, _Float16* __restrict__ out, int N) {
    int row = blockIdx.x;
    const float* x = X + (size_t)row * N;
    float s = 0.f, ss = 0.f;
    for (int c = threadIdx.x; c < N; c += blockDim.x) { float v = x[c]; s += v; ss += v * v; }
#pragma unroll
    for (int off = 16; off > 0; off >>= 1) {
        s  += __shfl_down(s, off, 32);
        ss += __shfl_down(ss, off, 32);
    }
    __shared__ float rs[8], rss[8];
    int lane = threadIdx.x & 31, wid = threadIdx.x >> 5;
    if (lane == 0) { rs[wid] = s; rss[wid] = ss; }
    __syncthreads();
    if (threadIdx.x == 0) {
        float S = 0.f, SS = 0.f;
        for (int i = 0; i < 8; ++i) { S += rs[i]; SS += rss[i]; }
        rs[0] = S; rss[0] = SS;
    }
    __syncthreads();
    float mu  = rs[0] / N;
    float var = rss[0] / N - mu * mu;
    float inv = rsqrtf(var + 1e-5f);
    for (int c = threadIdx.x; c < N; c += blockDim.x) {
        float v = (x[c] - mu) * inv * g[c] + bta[c];
        out[(size_t)row * N + c] = (_Float16)gelu_exact(v);
    }
}

// Gate head: logits(16) = Ghid(1024) @ W2 + b2; softmax; top-4 (desc, first-index
// tie-break like torch.topk); re-softmax over top-4. One block per token.
__global__ __launch_bounds__(128)
void gate2_topk_kernel(const _Float16* __restrict__ G, const float* __restrict__ W2,
                       const float* __restrict__ b2,
                       float* __restrict__ wsel, int* __restrict__ isel) {
    int row = blockIdx.x;
    __shared__ float sg[1024];
    __shared__ float slog[16];
    for (int c = threadIdx.x; c < 1024; c += 128) sg[c] = (float)G[(size_t)row * 1024 + c];
    __syncthreads();
    if (threadIdx.x < 16) {
        int e = threadIdx.x;
        float acc = b2[e];
        for (int k = 0; k < 1024; ++k) acc += sg[k] * W2[k * 16 + e];
        slog[e] = acc;
    }
    __syncthreads();
    if (threadIdx.x == 0) {
        float mx = slog[0];
        for (int i = 1; i < 16; ++i) mx = fmaxf(mx, slog[i]);
        float p[16], sum = 0.f;
        for (int i = 0; i < 16; ++i) { p[i] = expf(slog[i] - mx); sum += p[i]; }
        float invs = 1.0f / sum;
        for (int i = 0; i < 16; ++i) p[i] *= invs;
        int idx[4]; float val[4]; bool used[16];
        for (int i = 0; i < 16; ++i) used[i] = false;
        for (int k = 0; k < 4; ++k) {
            int bi = -1; float bv = -1.0f;
            for (int i = 0; i < 16; ++i)
                if (!used[i] && p[i] > bv) { bv = p[i]; bi = i; }
            used[bi] = true; idx[k] = bi; val[k] = bv;
        }
        float m2 = val[0];  // sorted descending
        float q[4], s2 = 0.f;
        for (int k = 0; k < 4; ++k) { q[k] = expf(val[k] - m2); s2 += q[k]; }
        float inv2 = 1.0f / s2;
        for (int k = 0; k < 4; ++k) {
            wsel[row * 4 + k] = q[k] * inv2;
            isel[row * 4 + k] = idx[k];
        }
    }
}

// Final head second GEMM: out[b, 0..19] = fhid[b, 0..511] @ W2 + b2
__global__ __launch_bounds__(64)
void final2_kernel(const _Float16* __restrict__ Xh, const float* __restrict__ W2,
                   const float* __restrict__ b2, float* __restrict__ out) {
    int row = blockIdx.x;
    __shared__ float sx[512];
    for (int c = threadIdx.x; c < 512; c += 64) sx[c] = (float)Xh[(size_t)row * 512 + c];
    __syncthreads();
    if (threadIdx.x < 20) {
        int c = threadIdx.x;
        float acc = b2[c];
        for (int k = 0; k < 512; ++k) acc += sx[k] * W2[k * 20 + c];
        out[(size_t)row * 20 + c] = acc;
    }
}

// ---------------------------------------------------------------------------
extern "C" void kernel_launch(void* const* d_in, const int* in_sizes, int n_in,
                              void* d_out, int out_size, void* d_ws, size_t ws_size,
                              hipStream_t stream) {
    const int B = 4096, D = 512, H = 1024, E = 16;

    const float* wifi    = (const float*)d_in[0];
    const float* rfid    = (const float*)d_in[1];
    const float* gate_w1 = (const float*)d_in[2];
    const float* gate_b1 = (const float*)d_in[3];
    const float* gln_g   = (const float*)d_in[4];
    const float* gln_b   = (const float*)d_in[5];
    const float* gate_w2 = (const float*)d_in[6];
    const float* gate_b2 = (const float*)d_in[7];
    const float* exp_w1  = (const float*)d_in[8];
    const float* exp_b1  = (const float*)d_in[9];
    const float* bn1_g   = (const float*)d_in[10];
    const float* bn1_b   = (const float*)d_in[11];
    const float* bn1_m   = (const float*)d_in[12];
    const float* bn1_v   = (const float*)d_in[13];
    const float* exp_w2  = (const float*)d_in[14];
    const float* exp_b2  = (const float*)d_in[15];
    const float* bn2_g   = (const float*)d_in[16];
    const float* bn2_b   = (const float*)d_in[17];
    const float* bn2_m   = (const float*)d_in[18];
    const float* bn2_v   = (const float*)d_in[19];
    const float* fin_w1  = (const float*)d_in[20];
    const float* fin_b1  = (const float*)d_in[21];
    const float* fbn_g   = (const float*)d_in[22];
    const float* fbn_b   = (const float*)d_in[23];
    const float* fbn_m   = (const float*)d_in[24];
    const float* fbn_v   = (const float*)d_in[25];
    const float* fin_w2  = (const float*)d_in[26];
    const float* fin_b2  = (const float*)d_in[27];
    float* out = (float*)d_out;

    // Bump-allocate workspace (all chunks 256B-aligned)
    char* p = (char*)d_ws;
    auto alloc = [&](size_t bytes) -> char* {
        uintptr_t a = ((uintptr_t)p + 255) & ~(uintptr_t)255;
        p = (char*)(a + bytes);
        return (char*)a;
    };
    _Float16* cmb16   = (_Float16*)alloc((size_t)B * D * 2);
    _Float16* gw1t    = (_Float16*)alloc((size_t)H * D * 2);       // [N=1024][K=512]
    float*    graw    = (float*)   alloc((size_t)B * H * 4);
    _Float16* ghid16  = (_Float16*)alloc((size_t)B * H * 2);
    float*    wsel    = (float*)   alloc((size_t)B * 4 * 4);
    int*      isel    = (int*)     alloc((size_t)B * 4 * 4);
    _Float16* ew1t    = (_Float16*)alloc((size_t)E * H * D * 2);   // per-e [N=1024][K=512]
    _Float16* ew2t    = (_Float16*)alloc((size_t)E * H * H * 2);   // per-e [N=1024][K=1024]
    _Float16* h1_16   = (_Float16*)alloc((size_t)B * H * 2);
    float*    fused   = (float*)   alloc((size_t)B * H * 4);
    _Float16* fused16 = (_Float16*)alloc((size_t)B * H * 2);
    _Float16* fw1t    = (_Float16*)alloc((size_t)D * H * 2);       // [N=512][K=1024]
    _Float16* fhid16  = (_Float16*)alloc((size_t)B * D * 2);
    (void)ws_size; (void)in_sizes; (void)n_in; (void)out_size;

    // 1) concat + f16 convert of activations
    concat_f16_kernel<<<(B * 512 + 255) / 256, 256, 0, stream>>>(wifi, rfid, cmb16, B);

    // 2) weight conversions (fp32 -> f16, transposed to [N][K])
    transpose_f16_kernel<<<((size_t)D * H + 255) / 256, 256, 0, stream>>>(gate_w1, gw1t, D, H, 1);
    transpose_f16_kernel<<<((size_t)E * D * H + 255) / 256, 256, 0, stream>>>(exp_w1, ew1t, D, H, E);
    transpose_f16_kernel<<<((size_t)E * H * H + 255) / 256, 256, 0, stream>>>(exp_w2, ew2t, H, H, E);
    transpose_f16_kernel<<<((size_t)H * D + 255) / 256, 256, 0, stream>>>(fin_w1, fw1t, H, D, 1);

    // 3) gate GEMM1: [B,512] x [512,1024] (+bias, fp32 out for LN)
    {
        dim3 grid(H / BN, B / BM);
        gemm_wmma_kernel<<<grid, 256, 0, stream>>>(cmb16, gw1t, B, H, D, EPI_BIAS_F32,
            gate_b1, nullptr, nullptr, nullptr, nullptr, graw, nullptr,
            nullptr, nullptr, 0, 0);
    }
    // 4) LayerNorm + GELU -> f16
    ln_gelu_kernel<<<B, 256, 0, stream>>>(graw, gln_g, gln_b, ghid16, H);
    // 5) gate GEMM2 (N=16) + softmax + top-4 + re-softmax
    gate2_topk_kernel<<<B, 128, 0, stream>>>(ghid16, gate_w2, gate_b2, wsel, isel);

    // 6) dense expert stack, sequential per expert (deterministic accumulation)
    for (int e = 0; e < E; ++e) {
        dim3 grid1(H / BN, B / BM);
        gemm_wmma_kernel<<<grid1, 256, 0, stream>>>(cmb16, ew1t + (size_t)e * H * D,
            B, H, D, EPI_BN_GELU_F16,
            exp_b1 + e * H, bn1_g + e * H, bn1_b + e * H, bn1_m + e * H, bn1_v + e * H,
            nullptr, h1_16, nullptr, nullptr, 0, 0);

        dim3 grid2(H / BN, B / BM);
        gemm_wmma_kernel<<<grid2, 256, 0, stream>>>(h1_16, ew2t + (size_t)e * H * H,
            B, H, H, EPI_BN_GELU_WACC,
            exp_b2 + e * H, bn2_g + e * H, bn2_b + e * H, bn2_m + e * H, bn2_v + e * H,
            fused, nullptr, wsel, isel, e, (e == 0) ? 1 : 0);
    }

    // 7) final head GEMM1: [B,1024] x [1024,512] + BN + GELU -> f16
    f32_to_f16_kernel<<<((size_t)B * H + 255) / 256, 256, 0, stream>>>(fused, fused16, (size_t)B * H);
    {
        dim3 grid(D / BN, B / BM);
        gemm_wmma_kernel<<<grid, 256, 0, stream>>>(fused16, fw1t, B, D, H, EPI_BN_GELU_F16,
            fin_b1, fbn_g, fbn_b, fbn_m, fbn_v, nullptr, fhid16,
            nullptr, nullptr, 0, 0);
    }
    // 8) final head GEMM2: [B,512] x [512,20] -> out
    final2_kernel<<<B, 64, 0, stream>>>(fhid16, fin_w2, fin_b2, out);
}
